// MultiheadAttention_2138893713699
// MI455X (gfx1250) — matrix operand order repaired and checked by reading, not compiled, not run on hardware
//
#include <hip/hip_runtime.h>
#include <hip/hip_bf16.h>
#include <math.h>

// ---------- problem constants ----------
#define BB    4
#define SS    2048
#define DM    1024
#define NH    16
#define HD    64
#define NQKV  3072   // 3*DM

typedef __attribute__((ext_vector_type(16))) __bf16 v16bf;
typedef __attribute__((ext_vector_type(8)))  float  v8f;

// ---------------------------------------------------------------------------
// Fragment-major indexing (CDNA5 ISA 7.12.2, wave32).
// A 16x32 bf16 fragment block = 512 elements, block[lane*16 + e]:
//   lane = ((k>>3)&1)*16 + m ; e = (k>>4)*8 + (k&7)
// B 32x16 bf16 fragment block = 512 elements, block[lane*16 + e]:
//   lane = ((k>>3)&1)*16 + n ; e = (k>>4)*8 + (k&7)
// ---------------------------------------------------------------------------
__device__ __forceinline__ int a_frag_idx(int m, int k) {
  return ((((k >> 3) & 1) << 4) + m) * 16 + ((k >> 4) << 3) + (k & 7);
}
__device__ __forceinline__ int b_frag_idx(int k, int n) {
  return ((((k >> 3) & 1) << 4) + n) * 16 + ((k >> 4) << 3) + (k & 7);
}

__device__ __forceinline__ v8f wmma_bf16(v16bf a, v16bf b, v8f c) {
  return __builtin_amdgcn_wmma_f32_16x16x32_bf16(
      false, a, false, b, (short)0, c, false, false);
}

__device__ __forceinline__ v16bf frag_ld(const __bf16* block) {
  const int lane = threadIdx.x & 31;
  return *(const v16bf*)(block + lane * 16);
}

// ---------------------------------------------------------------------------
// Kernel 1: qkv = x @ W_qkv + b_qkv -> bf16 Q/K/V in fragment-blocked global
// layouts.  Block = 128 threads (4 waves), 128x64 tile, wave owns 32x64.
// Because n0 is 64-aligned and 192 = 3*64, the (head, q/k/v) bucket is
// UNIFORM per block: head = blockIdx.y/3, which = blockIdx.y%3.
//
// Q layout: per (bh, 16-row tile t):  [2 kchunk frags][512]
// K layout: per (bh, 32-key chunk):   [4 frags: (key>>4)*2+(hd>>5)][512]
// V layout: per (bh, 32-key chunk):   [4 frags: hd>>4][512]
// ---------------------------------------------------------------------------
__global__ __launch_bounds__(128) void qkv_gemm_kernel(
    const float* __restrict__ x, const float* __restrict__ Wqkv,
    const float* __restrict__ bqkv,
    __bf16* __restrict__ Q, __bf16* __restrict__ K, __bf16* __restrict__ V) {
  __shared__ alignas(32) __bf16 As[8 * 512];   // 128 rows x 32 k
  __shared__ alignas(32) __bf16 Bs[4 * 512];   // 32 k x 64 n
  const int tid  = threadIdx.x;
  const int lane = tid & 31, warp = tid >> 5;
  const int m0 = blockIdx.x * 128;
  const int n0 = blockIdx.y * 64;

  const int a_col = tid & 31;                 // fixed k per thread
  const int a_kofs = ((a_col >> 4) << 3) + (a_col & 7);
  const int a_hofs = ((a_col >> 3) & 1) << 4;
  const int b_col = tid & 63;                 // fixed n per thread
  const int b_nofs = ((b_col >> 4) << 9) + ((b_col & 15) << 4);

  v8f acc[2][4] = {};
  for (int k0 = 0; k0 < DM; k0 += 32) {
#pragma unroll
    for (int i = 0; i < 32; ++i) {            // A: 4096 elems
      const int row = i * 4 + (tid >> 5);
      As[((row >> 4) << 9) + ((a_hofs + (row & 15)) << 4) + a_kofs] =
          (__bf16)x[(size_t)(m0 + row) * DM + k0 + a_col];
    }
#pragma unroll
    for (int i = 0; i < 16; ++i) {            // B: 2048 elems
      const int k = i * 2 + (tid >> 6);
      Bs[b_nofs + (((k >> 3) & 1) << 8) + ((k >> 4) << 3) + (k & 7)] =
          (__bf16)Wqkv[(size_t)(k0 + k) * NQKV + n0 + b_col];
    }
    __syncthreads();
    const v16bf a0 = frag_ld(&As[(warp * 2 + 0) * 512]);
    const v16bf a1 = frag_ld(&As[(warp * 2 + 1) * 512]);
#pragma unroll
    for (int cj = 0; cj < 4; ++cj) {
      const v16bf bfr = frag_ld(&Bs[cj * 512]);
      acc[0][cj] = wmma_bf16(a0, bfr, acc[0][cj]);
      acc[1][cj] = wmma_bf16(a1, bfr, acc[1][cj]);
    }
    __syncthreads();
  }

  // ---- uniform epilogue: one scalar branch, constant-folded swizzles ----
  const int head  = blockIdx.y / 3;           // block-uniform
  const int which = blockIdx.y % 3;           // block-uniform
  const int h = lane >> 4, nl = lane & 15;

  float bias[4];
#pragma unroll
  for (int cj = 0; cj < 4; ++cj) bias[cj] = bqkv[n0 + cj * 16 + nl];

#pragma unroll
  for (int ci = 0; ci < 2; ++ci) {
    const int gmBase = m0 + warp * 32 + ci * 16;     // 16-aligned row base
    const int bidx   = gmBase >> 11;
    const int s_tile = (gmBase & (SS - 1)) >> 4;     // 16-row tile index
    const size_t bh  = (size_t)(bidx * NH + head);
    if (which == 0) {
      __bf16* qt = Q + (bh * 128 + s_tile) * 1024;
#pragma unroll
      for (int cj = 0; cj < 4; ++cj)
#pragma unroll
        for (int r = 0; r < 8; ++r) {
          const int m = r + 8 * h, hd = cj * 16 + nl;
          qt[((hd >> 5) << 9) + a_frag_idx(m, hd & 31)] =
              (__bf16)(acc[ci][cj][r] + bias[cj]);
        }
    } else if (which == 1) {
      __bf16* kt = K + (bh * 64 + (s_tile >> 1)) * 2048;
#pragma unroll
      for (int cj = 0; cj < 4; ++cj)
#pragma unroll
        for (int r = 0; r < 8; ++r) {
          const int key = (s_tile & 1) * 16 + r + 8 * h;   // 0..31
          const int hd  = cj * 16 + nl;
          const int fb  = ((key >> 4) << 1) + (hd >> 5);
          kt[(fb << 9) + b_frag_idx(hd & 31, key & 15)] =
              (__bf16)(acc[ci][cj][r] + bias[cj]);
        }
    } else {
      __bf16* vt = V + (bh * 64 + (s_tile >> 1)) * 2048;
#pragma unroll
      for (int cj = 0; cj < 4; ++cj)
#pragma unroll
        for (int r = 0; r < 8; ++r) {
          const int key = (s_tile & 1) * 16 + r + 8 * h;
          const int hd  = cj * 16 + nl;
          vt[((hd >> 4) << 9) + b_frag_idx(key, hd & 15)] =
              (__bf16)(acc[ci][cj][r] + bias[cj]);
        }
    }
  }
}

// ---------------------------------------------------------------------------
// Kernel 2: flash attention.  grid = (S/64, B*H), block = 128 (4 waves).
// Each wave owns 16 query rows; all fragments are contiguous 32B/lane loads
// straight from pre-swizzled global Q/K/V.  No block barriers.
// ---------------------------------------------------------------------------
__global__ __launch_bounds__(128) void attn_kernel(
    const __bf16* __restrict__ Q, const __bf16* __restrict__ K,
    const __bf16* __restrict__ V, const float* __restrict__ mask,
    float* __restrict__ values) {
  __shared__ alignas(32) __bf16 Ps[4][512];

  const int tid  = threadIdx.x;
  const int lane = tid & 31, warp = tid >> 5;
  const int bh = blockIdx.y;
  const int b = bh >> 4, hh = bh & 15;
  const int q0 = blockIdx.x * 64 + warp * 16;
  const int hlf = lane >> 4, nl = lane & 15;

  const __bf16* qt = Q + ((size_t)bh * 128 + (q0 >> 4)) * 1024;
  const v16bf qa0 = frag_ld(qt);
  const v16bf qa1 = frag_ld(qt + 512);

  v8f o[4] = {};
  float rmax[8], rsum[8];
#pragma unroll
  for (int r = 0; r < 8; ++r) { rmax[r] = -INFINITY; rsum[r] = 0.f; }

  const __bf16* kbh = K + (size_t)bh * 64 * 2048;
  const __bf16* vbh = V + (size_t)bh * 64 * 2048;

  for (int ch = 0; ch < 64; ++ch) {           // 32 keys per chunk
    const __bf16* kcb = kbh + (size_t)ch * 2048;
    const __bf16* vcb = vbh + (size_t)ch * 2048;
    if (ch + 1 < 64) {                        // global_prefetch_b8
      __builtin_prefetch(kcb + 2048, 0, 1);
      __builtin_prefetch(vcb + 2048, 0, 1);
    }

    v8f s0 = {}, s1 = {};
    s0 = wmma_bf16(qa0, frag_ld(kcb + 0 * 512), s0);
    s0 = wmma_bf16(qa1, frag_ld(kcb + 1 * 512), s0);
    s1 = wmma_bf16(qa0, frag_ld(kcb + 2 * 512), s1);
    s1 = wmma_bf16(qa1, frag_ld(kcb + 3 * 512), s1);

    const int kc = ch * 32;
    const float scale = 0.125f;               // 1/sqrt(64)
    float p0[8], p1[8], alpha[8];
#pragma unroll
    for (int r = 0; r < 8; ++r) {
      const int qrow = q0 + r + 8 * hlf;
      const float* mrow = mask + (size_t)qrow * SS + kc;
      float v0 = s0[r] * scale + mrow[nl];
      float v1 = s1[r] * scale + mrow[nl + 16];
      float mx = fmaxf(v0, v1);
      for (int off = 1; off < 16; off <<= 1)
        mx = fmaxf(mx, __shfl_xor(mx, off, 32));
      const float mnew = fmaxf(rmax[r], mx);
      const float a  = __expf(rmax[r] - mnew);
      const float e0 = __expf(v0 - mnew);
      const float e1 = __expf(v1 - mnew);
      float cs = e0 + e1;
      for (int off = 1; off < 16; off <<= 1)
        cs += __shfl_xor(cs, off, 32);
      rsum[r] = rsum[r] * a + cs;
      rmax[r] = mnew;
      alpha[r] = a;
      p0[r] = e0; p1[r] = e1;
    }

#pragma unroll
    for (int sl = 0; sl < 4; ++sl)
#pragma unroll
      for (int r = 0; r < 8; ++r) o[sl][r] *= alpha[r];

    // P -> per-wave LDS fragment block (scatter), then contiguous reload
#pragma unroll
    for (int r = 0; r < 8; ++r) {
      const int row = r + 8 * hlf;
      Ps[warp][a_frag_idx(row, nl)]      = (__bf16)p0[r];
      Ps[warp][a_frag_idx(row, nl + 16)] = (__bf16)p1[r];
    }
    const v16bf pa = frag_ld(Ps[warp]);
#pragma unroll
    for (int sl = 0; sl < 4; ++sl)
      o[sl] = wmma_bf16(pa, frag_ld(vcb + sl * 512), o[sl]);
  }

#pragma unroll
  for (int r = 0; r < 8; ++r) rsum[r] = 1.f / rsum[r];
#pragma unroll
  for (int sl = 0; sl < 4; ++sl)
#pragma unroll
    for (int r = 0; r < 8; ++r) {
      const int qrow = q0 + r + 8 * hlf;
      const int col = hh * HD + sl * 16 + nl;
      values[((size_t)b * SS + qrow) * DM + col] = o[sl][r] * rsum[r];
    }
}

// ---------------------------------------------------------------------------
// Kernel 3: out = values @ W_o + b_o  (values fp32 read from d_out)
// ---------------------------------------------------------------------------
__global__ __launch_bounds__(128) void out_gemm_kernel(
    const float* __restrict__ Vin, const float* __restrict__ Wo,
    const float* __restrict__ bo, float* __restrict__ out) {
  __shared__ alignas(32) __bf16 As[8 * 512];
  __shared__ alignas(32) __bf16 Bs[4 * 512];
  const int tid  = threadIdx.x;
  const int lane = tid & 31, warp = tid >> 5;
  const int m0 = blockIdx.x * 128;
  const int n0 = blockIdx.y * 64;

  const int a_col = tid & 31;
  const int a_kofs = ((a_col >> 4) << 3) + (a_col & 7);
  const int a_hofs = ((a_col >> 3) & 1) << 4;
  const int b_col = tid & 63;
  const int b_nofs = ((b_col >> 4) << 9) + ((b_col & 15) << 4);

  v8f acc[2][4] = {};
  for (int k0 = 0; k0 < DM; k0 += 32) {
#pragma unroll
    for (int i = 0; i < 32; ++i) {
      const int row = i * 4 + (tid >> 5);
      As[((row >> 4) << 9) + ((a_hofs + (row & 15)) << 4) + a_kofs] =
          (__bf16)Vin[(size_t)(m0 + row) * DM + k0 + a_col];
    }
#pragma unroll
    for (int i = 0; i < 16; ++i) {
      const int k = i * 2 + (tid >> 6);
      Bs[b_nofs + (((k >> 3) & 1) << 8) + ((k >> 4) << 3) + (k & 7)] =
          (__bf16)Wo[(size_t)(k0 + k) * DM + n0 + b_col];
    }
    __syncthreads();
    const v16bf a0 = frag_ld(&As[(warp * 2 + 0) * 512]);
    const v16bf a1 = frag_ld(&As[(warp * 2 + 1) * 512]);
#pragma unroll
    for (int cj = 0; cj < 4; ++cj) {
      const v16bf bfr = frag_ld(&Bs[cj * 512]);
      acc[0][cj] = wmma_bf16(a0, bfr, acc[0][cj]);
      acc[1][cj] = wmma_bf16(a1, bfr, acc[1][cj]);
    }
    __syncthreads();
  }

  const int h = lane >> 4, nl = lane & 15;
  float bias[4];
#pragma unroll
  for (int cj = 0; cj < 4; ++cj) bias[cj] = bo[n0 + cj * 16 + nl];

#pragma unroll
  for (int ci = 0; ci < 2; ++ci)
#pragma unroll
    for (int cj = 0; cj < 4; ++cj)
#pragma unroll
      for (int r = 0; r < 8; ++r) {
        const int gm = m0 + warp * 32 + ci * 16 + r + 8 * h;
        const int gn = n0 + cj * 16 + nl;
        out[(size_t)gm * DM + gn] = acc[ci][cj][r] + bias[cj];
      }
}

// ---------------------------------------------------------------------------
extern "C" void kernel_launch(void* const* d_in, const int* in_sizes, int n_in,
                              void* d_out, int out_size, void* d_ws, size_t ws_size,
                              hipStream_t stream) {
  const float* x    = (const float*)d_in[0];
  const float* mask = (const float*)d_in[1];
  const float* Wqkv = (const float*)d_in[2];
  const float* bqkv = (const float*)d_in[3];
  const float* Wo   = (const float*)d_in[4];
  const float* bo   = (const float*)d_in[5];

  float* out    = (float*)d_out;                        // [B,S,DM]
  float* values = out + (size_t)BB * SS * DM;           // second tuple output

  const size_t headElems = (size_t)BB * NH * SS * HD;   // 8,388,608
  __bf16* Q = (__bf16*)d_ws;
  __bf16* K = Q + headElems;
  __bf16* V = K + headElems;                            // 48 MB total in ws

  qkv_gemm_kernel<<<dim3((BB * SS) / 128, NQKV / 64), 128, 0, stream>>>(
      x, Wqkv, bqkv, Q, K, V);
  attn_kernel<<<dim3(SS / 64, BB * NH), 128, 0, stream>>>(
      Q, K, V, mask, values);
  out_gemm_kernel<<<dim3((BB * SS) / 128, DM / 64), 128, 0, stream>>>(
      values, Wo, bo, out);
}
